// BinaryHopfieldNetwork_71098888618577
// MI455X (gfx1250) — matrix operand order repaired and checked by reading, not compiled, not run on hardware
//
#include <hip/hip_runtime.h>

#define NN 8192
#define STEPS 512

typedef __attribute__((ext_vector_type(16))) _Float16 v16h;
typedef __attribute__((ext_vector_type(8)))  float    v8f;

// ---------------------------------------------------------------------------
// Phase 1: h = W @ x0 via v_wmma_f32_16x16x32_f16 (exact: W,x in {-1,0,1}).
// Block = 128 threads = 4 waves. Block b owns rows [16b, 16b+16).
// Wave w accumulates K-quarter [2048w, 2048(w+1)) in 64 WMMAs, partials
// reduced through LDS. W is read exactly once (bandwidth-bound).
// ---------------------------------------------------------------------------
__global__ __launch_bounds__(128)
void hopfield_matvec_wmma(const float* __restrict__ W,
                          const float* __restrict__ x,
                          float* __restrict__ h)
{
    __shared__ _Float16 xs[NN];        // 16 KB: x0 as f16 (exact)
    __shared__ float    hpart[4][16];

    const int tid = threadIdx.x;
    for (int j = tid; j < NN; j += 128) xs[j] = (_Float16)x[j];
    __syncthreads();

    const int wave = tid >> 5;         // K-quarter 0..3
    const int lane = tid & 31;
    const int m    = lane & 15;        // row within strip
    const int hi   = lane >> 4;        // lane-half per ISA A/B layouts
    const int row  = blockIdx.x * 16 + m;

    const float*    rowp = W + (size_t)row * NN + wave * 2048;
    const _Float16* xq   = xs + wave * 2048;

    v8f acc = {};
    for (int it = 0; it < 64; ++it) {
        const int kb = it * 32;
        // A tile (16x32 f16): lanes 0-15 hold K kb+0..7 / kb+16..23,
        //                     lanes 16-31 hold K kb+8..15 / kb+24..31.
        const float4 f0 = *(const float4*)(rowp + kb + hi * 8);
        const float4 f1 = *(const float4*)(rowp + kb + hi * 8 + 4);
        const float4 f2 = *(const float4*)(rowp + kb + 16 + hi * 8);
        const float4 f3 = *(const float4*)(rowp + kb + 16 + hi * 8 + 4);
        v16h a;
        a[0]=(_Float16)f0.x;  a[1]=(_Float16)f0.y;  a[2]=(_Float16)f0.z;  a[3]=(_Float16)f0.w;
        a[4]=(_Float16)f1.x;  a[5]=(_Float16)f1.y;  a[6]=(_Float16)f1.z;  a[7]=(_Float16)f1.w;
        a[8]=(_Float16)f2.x;  a[9]=(_Float16)f2.y;  a[10]=(_Float16)f2.z; a[11]=(_Float16)f2.w;
        a[12]=(_Float16)f3.x; a[13]=(_Float16)f3.y; a[14]=(_Float16)f3.z; a[15]=(_Float16)f3.w;
        // B tile (32x16 f16): every column = x slice (lanes 0-15 K kb+0..7 / kb+16..23,
        // lanes 16-31 K kb+8..15 / kb+24..31). Only D column 0 is consumed; making all
        // columns identical avoids lane divergence.
        const _Float16* xp = xq + kb + hi * 8;
        v16h b;
        #pragma unroll
        for (int j = 0; j < 16; ++j) b[j] = xp[j < 8 ? j : j + 8];

        acc = __builtin_amdgcn_wmma_f32_16x16x32_f16(
                  false, a, false, b, (short)0, acc, false, false);
    }

    // D layout: lane 0 holds (M=0..7, N=0) in acc[0..7]; lane 16 holds M=8..15.
    if (lane == 0) {
        #pragma unroll
        for (int r = 0; r < 8; ++r) hpart[wave][r] = acc[r];
    }
    if (lane == 16) {
        #pragma unroll
        for (int r = 0; r < 8; ++r) hpart[wave][8 + r] = acc[r];
    }
    __syncthreads();
    if (tid < 16)
        h[blockIdx.x * 16 + tid] =
            hpart[0][tid] + hpart[1][tid] + hpart[2][tid] + hpart[3][tid];
}

// ---------------------------------------------------------------------------
// Phase 2: single 1024-thread workgroup runs the 512 sequential steps with
// exact integer bookkeeping. h and x live in LDS; flips do one contiguous
// 32 KB row read (prefetched into L2 up front via global_prefetch_b8).
// ---------------------------------------------------------------------------
__global__ __launch_bounds__(1024)
void hopfield_steps(const float* __restrict__ W,
                    const float* __restrict__ x_noisy,
                    const int*   __restrict__ idx,
                    const float* __restrict__ hf,
                    float* __restrict__ out)
{
    __shared__ int       hs[NN];       // 32 KB
    __shared__ int       xsI[NN];      // 32 KB
    __shared__ long long red[1024];    //  8 KB

    const int tid = threadIdx.x;

    // Load state and form partial S = x . h  (all exact integers).
    long long part = 0;
    for (int j = tid; j < NN; j += 1024) {
        const int hv = __float2int_rn(hf[j]);
        const float xv = x_noisy[j];
        const int xi = (xv > 0.f) ? 1 : ((xv < 0.f) ? -1 : 0);
        hs[j]  = hv;
        xsI[j] = xi;
        part += (long long)xi * hv;
    }

    // Prefetch all candidate rows (16 MB total, fits easily in 192 MB L2).
    for (int s = 0; s < STEPS; ++s) {
        const float* rowp = W + (size_t)idx[s] * NN;
        const int j = tid * 32;                 // 128 B per prefetch
        if (j < NN) __builtin_prefetch(rowp + j, 0, 3);
    }

    red[tid] = part;
    __syncthreads();
    for (int off = 512; off > 0; off >>= 1) {
        if (tid < off) red[tid] += red[tid + off];
        __syncthreads();
    }
    long long S = red[0];                       // every thread tracks S

    for (int s = 0; s < STEPS; ++s) {
        __syncthreads();                        // previous h update visible
        const int i   = idx[s];
        const int hio = hs[i];
        const int xi  = xsI[i];
        const int act = (hio > 0) - (hio < 0);
        const int sx  = (xi  > 0) - (xi  < 0);
        const int delta = (sx != act) ? (act - xi) : 0;   // uniform

        if (delta != 0) {
            __syncthreads();                    // everyone read hio/xi first
            const float* rowp = W + (size_t)i * NN;
            const int wii = __float2int_rn(rowp[i]);      // broadcast load
            #pragma unroll
            for (int t = 0; t < 8; ++t) {
                const int j = tid + t * 1024;             // coalesced 4 KB
                hs[j] += delta * __float2int_rn(rowp[j]);
            }
            if (tid == 0) xsI[i] = xi + delta;            // == act
            S += (long long)(2 * delta) * hio
               + (long long)(delta * delta) * wii;
        }
        if (tid == 0) out[NN + s] = -0.5f * (float)S;
    }

    __syncthreads();
    for (int j = tid; j < NN; j += 1024) out[j] = (float)xsI[j];
}

extern "C" void kernel_launch(void* const* d_in, const int* in_sizes, int n_in,
                              void* d_out, int out_size, void* d_ws, size_t ws_size,
                              hipStream_t stream)
{
    const float* x_noisy = (const float*)d_in[0];   // 8192 fp32
    const float* weights = (const float*)d_in[1];   // 8192*8192 fp32
    const int*   idx     = (const int*)d_in[2];     // 512 int32
    float* out = (float*)d_out;                     // [x_final(8192) | energies(512)]
    float* h   = (float*)d_ws;                      // 32 KB scratch: h = W @ x0

    hopfield_matvec_wmma<<<NN / 16, 128, 0, stream>>>(weights, x_noisy, h);
    hopfield_steps<<<1, 1024, 0, stream>>>(weights, x_noisy, idx, h, out);
}
